// NeuralODE_58016418235029
// MI455X (gfx1250) — matrix operand order repaired and checked by reading, not compiled, not run on hardware
//
#include <hip/hip_runtime.h>

typedef __attribute__((ext_vector_type(16))) _Float16 v16h;
typedef __attribute__((ext_vector_type(8)))  float    v8f;

namespace {
constexpr int NROW = 16;     // trajectories per workgroup
constexpr int STR  = 68;     // padded LDS row stride (floats), bank-spreading
constexpr int MAXS = 256;

// Tsit5 tableau
constexpr float cA21 = 0.161f;
constexpr float cA31 = -0.008480655492356989f, cA32 = 0.335480655492357f;
constexpr float cA41 = 2.8971530571054935f,  cA42 = -6.359448489975075f,  cA43 = 4.3622954328695815f;
constexpr float cA51 = 5.325864828439257f,   cA52 = -11.748883564062828f, cA53 = 7.4955393428898365f, cA54 = -0.09249506636175525f;
constexpr float cA61 = 5.86145544294642f,    cA62 = -12.92096931784711f,  cA63 = 8.159367898576159f,  cA64 = -0.071584973281401f, cA65 = -0.028269050394068383f;
constexpr float cB1 = 0.09646076681806523f, cB2 = 0.01f, cB3 = 0.4798896504144996f;
constexpr float cB4 = 1.379008574103742f,   cB5 = -3.290069515436081f, cB6 = 2.324710524099774f;
constexpr float cE1 = -0.00178001105222577714f, cE2 = -0.0008164344596567469f, cE3 = 0.007880878010261995f;
constexpr float cE4 = -0.1447110071732629f,     cE5 = 0.5823571654525552f;
constexpr float cE6 = -0.45808210592918697f,    cE7 = 0.015151515151515152f;
}

struct Frag { v16h h0, l0, h1, l1; };   // B-fragments: K-chunk 0/1, hi/lo split

__device__ __forceinline__ float softplusf(float x) {
    return fmaxf(x, 0.0f) + log1pf(expf(-fabsf(x)));
}

// Pack a 64x64 row-major weight matrix's B-fragments for this wave's N-tile.
// B (32x16) layout: lanes 0-15 -> N=lane, K=0..15 ; lanes 16-31 -> N=lane-16, K=16..31.
__device__ __forceinline__ Frag load_frag(const float* __restrict__ W, int lane, int wave) {
    Frag f;
    const int n  = wave * 16 + (lane & 15);
    const int kb = (lane & 16) ? 16 : 0;
    #pragma unroll
    for (int i = 0; i < 16; ++i) {
        float x0 = W[n * 64 + kb + i];        // K-chunk 0 (k = kb+i)
        float x1 = W[n * 64 + 32 + kb + i];   // K-chunk 1 (k = 32+kb+i)
        _Float16 h0 = (_Float16)x0;
        _Float16 h1 = (_Float16)x1;
        f.h0[i] = h0; f.l0[i] = (_Float16)(x0 - (float)h0);
        f.h1[i] = h1; f.l1[i] = (_Float16)(x1 - (float)h1);
    }
    return f;
}

// One dense layer: OUT[16x64 tile-col] = act(IN[16x64] * W^T + b), via hi/lo split WMMA.
// A (16x32 f16) layout: lane -> M=lane&15; K offsets 0..15 (+16 for upper half-wave).
template<bool SP>
__device__ __forceinline__ void layer(const float* IN, float* OUT,
                                      const Frag& wf, float bias,
                                      int lane, int wave) {
    __syncthreads();                       // IN fully produced by all threads
    const int m  = lane & 15;
    const int ka = (lane & 16) ? 16 : 0;
    v8f acc = {0.f, 0.f, 0.f, 0.f, 0.f, 0.f, 0.f, 0.f};
    #pragma unroll
    for (int kc = 0; kc < 2; ++kc) {
        const float* row = IN + m * STR + kc * 32 + ka;
        v16h ah, al;
        #pragma unroll
        for (int i = 0; i < 16; ++i) {
            float x = row[i];
            _Float16 hi = (_Float16)x;
            ah[i] = hi;
            al[i] = (_Float16)(x - (float)hi);
        }
        v16h bh = kc ? wf.h1 : wf.h0;
        v16h bl = kc ? wf.l1 : wf.l0;
        acc = __builtin_amdgcn_wmma_f32_16x16x32_f16(false, ah, false, bh, (short)0, acc, false, false);
        acc = __builtin_amdgcn_wmma_f32_16x16x32_f16(false, al, false, bh, (short)0, acc, false, false);
        acc = __builtin_amdgcn_wmma_f32_16x16x32_f16(false, ah, false, bl, (short)0, acc, false, false);
    }
    const int n  = wave * 16 + (lane & 15);
    const int mb = (lane & 16) ? 8 : 0;
    #pragma unroll
    for (int r = 0; r < 8; ++r) {
        float v = acc[r] + bias;
        if (SP) v = softplusf(v);
        OUT[(mb + r) * STR + n] = v;
    }
}

__global__ __launch_bounds__(128) void node_tsit5_kernel(
    const float* __restrict__ ts, const float* __restrict__ y0,
    const float* __restrict__ w1, const float* __restrict__ b1,
    const float* __restrict__ w2, const float* __restrict__ b2,
    const float* __restrict__ w3, const float* __restrict__ b3,
    float* __restrict__ out, int T)
{
    __shared__ float sY [NROW * STR], sF [NROW * STR];
    __shared__ float sK2[NROW * STR], sK3[NROW * STR], sK4[NROW * STR];
    __shared__ float sK5[NROW * STR], sK6[NROW * STR], sK7[NROW * STR];
    __shared__ float sYN[NROW * STR], sX [NROW * STR], sH1[NROW * STR];
    __shared__ float t_s[16], h_s[16], heff_s[16], told_s[16], tnew_s[16];
    __shared__ float part_s[16][8];
    __shared__ int   cur_s[16], acc_s[16], done_s[16], alldone_s;

    const int tid  = threadIdx.x;
    const int lane = tid & 31;
    const int wave = tid >> 5;
    const int base = blockIdx.x * NROW;
    const int em   = tid >> 3;          // row handled by this thread (elementwise work)
    const int sub  = tid & 7;           // 8 threads/row, 8 contiguous dims each
    const int ed   = sub * 8;

    const float t0  = ts[0];
    const float t1v = ts[T - 1];
    const float dt0 = ts[1] - ts[0];

    // Persistent weight fragments + per-lane bias (n fixed per lane/wave).
    const Frag f1 = load_frag(w1, lane, wave);
    const Frag f2 = load_frag(w2, lane, wave);
    const Frag f3 = load_frag(w3, lane, wave);
    const int  ncol = wave * 16 + (lane & 15);
    const float bias1 = b1[ncol], bias2 = b2[ncol], bias3 = b3[ncol];

    // Load initial state tile
    #pragma unroll
    for (int i = 0; i < 8; ++i)
        sY[em * STR + ed + i] = y0[(base + em) * 64 + ed + i];
    if (tid < 16) { t_s[tid] = t0; h_s[tid] = dt0; cur_s[tid] = 0; }
    __syncthreads();

    // Emit save points at/below t0 (== y0)
    {
        int j = 0;
        while (j < T && ts[j] <= t0) {
            #pragma unroll
            for (int i = 0; i < 8; ++i)
                out[((size_t)(base + em) * T + j) * 64 + ed + i] = sY[em * STR + ed + i];
            ++j;
        }
        if (sub == 0) cur_s[em] = j;
    }

    // Full MLP eval (layer2 reuses sX as scratch; safe even when IN==sX thanks to entry barriers)
    auto mlp = [&](const float* IN, float* OUT) {
        layer<true >(IN,  sH1, f1, bias1, lane, wave);
        layer<true >(sH1, sX,  f2, bias2, lane, wave);
        layer<false>(sX,  OUT, f3, bias3, lane, wave);
    };

    mlp(sY, sF);   // f0 = vf(y0)  (FSAL seed)

    for (int it = 0; it < MAXS; ++it) {
        __syncthreads();
        if (tid < 16) {
            float t = t_s[tid];
            done_s[tid] = ((t1v - t) <= 1e-10f * (t1v - t0)) ? 1 : 0;
            heff_s[tid] = fmaxf(fminf(h_s[tid], t1v - t), 0.0f);
        }
        __syncthreads();
        if (tid == 0) {
            int ad = 1;
            for (int m = 0; m < 16; ++m) ad &= done_s[m];
            alldone_s = ad;
        }
        __syncthreads();
        if (alldone_s) break;

        const float he = heff_s[em];

        // ---- stage combinations + vf evals (k1 = sF) ----
        __syncthreads();
        #pragma unroll
        for (int i = 0; i < 8; ++i) { int o = em * STR + ed + i;
            sX[o] = sY[o] + he * (cA21 * sF[o]); }
        mlp(sX, sK2);

        __syncthreads();
        #pragma unroll
        for (int i = 0; i < 8; ++i) { int o = em * STR + ed + i;
            sX[o] = sY[o] + he * (cA31 * sF[o] + cA32 * sK2[o]); }
        mlp(sX, sK3);

        __syncthreads();
        #pragma unroll
        for (int i = 0; i < 8; ++i) { int o = em * STR + ed + i;
            sX[o] = sY[o] + he * (cA41 * sF[o] + cA42 * sK2[o] + cA43 * sK3[o]); }
        mlp(sX, sK4);

        __syncthreads();
        #pragma unroll
        for (int i = 0; i < 8; ++i) { int o = em * STR + ed + i;
            sX[o] = sY[o] + he * (cA51 * sF[o] + cA52 * sK2[o] + cA53 * sK3[o] + cA54 * sK4[o]); }
        mlp(sX, sK5);

        __syncthreads();
        #pragma unroll
        for (int i = 0; i < 8; ++i) { int o = em * STR + ed + i;
            sX[o] = sY[o] + he * (cA61 * sF[o] + cA62 * sK2[o] + cA63 * sK3[o] + cA64 * sK4[o] + cA65 * sK5[o]); }
        mlp(sX, sK6);

        __syncthreads();
        #pragma unroll
        for (int i = 0; i < 8; ++i) { int o = em * STR + ed + i;
            sYN[o] = sY[o] + he * (cB1 * sF[o] + cB2 * sK2[o] + cB3 * sK3[o]
                                 + cB4 * sK4[o] + cB5 * sK5[o] + cB6 * sK6[o]); }
        mlp(sYN, sK7);

        // ---- embedded error norm (per row) ----
        __syncthreads();
        {
            float s = 0.0f;
            #pragma unroll
            for (int i = 0; i < 8; ++i) { int o = em * STR + ed + i;
                float e  = he * (cE1 * sF[o] + cE2 * sK2[o] + cE3 * sK3[o] + cE4 * sK4[o]
                               + cE5 * sK5[o] + cE6 * sK6[o] + cE7 * sK7[o]);
                float sc = 1e-6f + 1e-3f * fmaxf(fabsf(sY[o]), fabsf(sYN[o]));
                float r  = e / sc;
                s += r * r;
            }
            part_s[em][sub] = s;
        }
        __syncthreads();
        if (tid < 16) {
            float s = 0.0f;
            #pragma unroll
            for (int i = 0; i < 8; ++i) s += part_s[tid][i];
            float errn   = sqrtf(s * (1.0f / 64.0f));
            int   dn     = done_s[tid];
            int   accept = (errn <= 1.0f) ? 1 : 0;
            float factor = 0.9f * powf(errn + 1e-10f, -0.2f);
            factor = fminf(fmaxf(factor, 0.2f), 5.0f);
            float hef = heff_s[tid];
            told_s[tid] = t_s[tid];
            if (!dn) {
                if (accept) t_s[tid] = t_s[tid] + hef;
                h_s[tid] = hef * factor;
            }
            tnew_s[tid] = t_s[tid];
            acc_s[tid]  = (accept && !dn) ? 1 : 0;
        }
        __syncthreads();

        // ---- online interpolation of save points in (t_old, t_new] ----
        if (acc_s[em]) {
            float tl = told_s[em], tr = tnew_s[em];
            float inv = 1.0f / (tr - tl);
            int j = cur_s[em];
            while (j < T) {
                float tj = ts[j];
                if (tj > tr) break;
                float th = (tj - tl) * inv;
                #pragma unroll
                for (int i = 0; i < 8; ++i) { int o = em * STR + ed + i;
                    float yl = sY[o], yr = sYN[o];
                    out[((size_t)(base + em) * T + j) * 64 + ed + i] = yl + th * (yr - yl);
                }
                ++j;
            }
            if (sub == 0) cur_s[em] = j;
        }
        __syncthreads();

        // ---- state update (accepted & not done rows): y <- y_new, f <- k7 (FSAL) ----
        if (acc_s[em]) {
            #pragma unroll
            for (int i = 0; i < 8; ++i) { int o = em * STR + ed + i;
                sY[o] = sYN[o];
                sF[o] = sK7[o];
            }
        }
    }

    // ---- tail: flush any un-emitted save points with the final state ----
    __syncthreads();
    {
        int j = cur_s[em];
        while (j < T) {
            #pragma unroll
            for (int i = 0; i < 8; ++i)
                out[((size_t)(base + em) * T + j) * 64 + ed + i] = sY[em * STR + ed + i];
            ++j;
        }
    }
}

extern "C" void kernel_launch(void* const* d_in, const int* in_sizes, int n_in,
                              void* d_out, int out_size, void* d_ws, size_t ws_size,
                              hipStream_t stream) {
    (void)n_in; (void)out_size; (void)d_ws; (void)ws_size;
    const float* ts = (const float*)d_in[0];
    const float* y0 = (const float*)d_in[1];
    const float* w1 = (const float*)d_in[2];
    const float* b1 = (const float*)d_in[3];
    const float* w2 = (const float*)d_in[4];
    const float* b2 = (const float*)d_in[5];
    const float* w3 = (const float*)d_in[6];
    const float* b3 = (const float*)d_in[7];
    float* out = (float*)d_out;

    const int T = in_sizes[0];
    const int B = in_sizes[1] / 64;          // 512
    const int grid = B / NROW;               // 32 workgroups, 4 waves each

    node_tsit5_kernel<<<grid, 128, 0, stream>>>(ts, y0, w1, b1, w2, b2, w3, b3, out, T);
}
